// HyperMoMixLinear_45981919871029
// MI455X (gfx1250) — matrix operand order repaired and checked by reading, not compiled, not run on hardware
//
#include <hip/hip_runtime.h>
#include <cstdint>
#include <cstddef>

// ---------------------------------------------------------------------------
// HyperMoMixLinear for MI455X (gfx1250, wave32, WMMA)
//   B=4 S=128 -> T=512 tokens, IN=OUT=1024, C=256, K=4, NI=NO=32, D1=D2=32768
// Dominant work: two [512,1024]@[1024,32768] GEMMs -> v_wmma_f32_16x16x32_bf16,
// fused with mixed-bias epilogue + per-token Monarch contraction (no M1/M2
// materialization, no atomics). Weights stay f32 in HBM (read once, ~268MB),
// converted to bf16 while staging into LDS. 133KB LDS/WG -> 2 WG per WGP so
// staging of one workgroup overlaps WMMA of the other.
// ---------------------------------------------------------------------------

#define T_TOKENS 512
#define CDIM     256
#define KEXP     4
#define DBIG     32768
#define HC       1024   // K*C : folded hyper-network K-dimension

typedef __attribute__((ext_vector_type(16))) __bf16 v16bf;
typedef __attribute__((ext_vector_type(8)))  __bf16 v8bf;
typedef __attribute__((ext_vector_type(8)))  float  v8f;

__device__ __forceinline__ float silu_f(float v) { return v / (1.0f + __expf(-v)); }

// ---------------------------------------------------------------------------
// Kernel 1: per-token prep.  c = silu(x@Wc+bc); mixer softmax coeffs;
// hcat{1,2}[t, k*256+j] = bf16(coeffs[t,k] * silu(c@W{1,2}a[k] + b{1,2}a[k]));
// ratio = c@Wr+br; ybias = c@Wb+bb.
// ---------------------------------------------------------------------------
__global__ __launch_bounds__(256) void hmx_prep(
    const float* __restrict__ x,   const float* __restrict__ Wc,  const float* __restrict__ bc,
    const float* __restrict__ W1a, const float* __restrict__ b1a,
    const float* __restrict__ W2a, const float* __restrict__ b2a,
    const float* __restrict__ Wm1, const float* __restrict__ bm1,
    const float* __restrict__ Wm2, const float* __restrict__ bm2,
    const float* __restrict__ Wb,  const float* __restrict__ bb,
    const float* __restrict__ Wr,  const float* __restrict__ br,
    __bf16* __restrict__ hcat1, __bf16* __restrict__ hcat2,
    float* __restrict__ coeffs, float* __restrict__ ratio, float* __restrict__ ybias)
{
    __shared__ float xt[1024];
    __shared__ float cbuf[CDIM];
    __shared__ float mbuf[CDIM];
    __shared__ float coef[KEXP];

    const int t = blockIdx.x;
    const int j = threadIdx.x;

    const float* xr = x + (size_t)t * 1024;
    for (int e = j; e < 1024; e += 256) xt[e] = xr[e];
    __syncthreads();

    // compressor: c[j] = silu(sum_i x[i] * Wc[i,j] + bc[j])   (coalesced over j)
    float a = bc[j];
    for (int i = 0; i < 1024; ++i) a = fmaf(xt[i], Wc[i * CDIM + j], a);
    cbuf[j] = silu_f(a);
    __syncthreads();

    // mixer hidden
    float m = bm1[j];
    for (int i = 0; i < CDIM; ++i) m = fmaf(cbuf[i], Wm1[i * CDIM + j], m);
    mbuf[j] = silu_f(m);
    __syncthreads();

    if (j == 0) {
        float lg[KEXP];
        float mx = -1e30f;
        for (int k = 0; k < KEXP; ++k) {
            float s = bm2[k];
            for (int i = 0; i < CDIM; ++i) s = fmaf(mbuf[i], Wm2[i * KEXP + k], s);
            lg[k] = s; mx = fmaxf(mx, s);
        }
        float den = 0.f;
        for (int k = 0; k < KEXP; ++k) { lg[k] = __expf(lg[k] - mx); den += lg[k]; }
        for (int k = 0; k < KEXP; ++k) {
            float cc = lg[k] / den;
            coef[k] = cc;
            coeffs[t * KEXP + k] = cc;
        }
        float r = br[0];
        for (int i = 0; i < CDIM; ++i) r = fmaf(cbuf[i], Wr[i], r);
        ratio[t] = r;
    }
    __syncthreads();

    // expert hidden states, pre-scaled by mixing coefficients, stored bf16
    for (int k = 0; k < KEXP; ++k) {
        float a1 = b1a[k * CDIM + j];
        float a2 = b2a[k * CDIM + j];
        const float* w1 = W1a + (size_t)k * (CDIM * CDIM) + j;
        const float* w2 = W2a + (size_t)k * (CDIM * CDIM) + j;
        for (int i = 0; i < CDIM; ++i) {
            a1 = fmaf(cbuf[i], w1[i * CDIM], a1);
            a2 = fmaf(cbuf[i], w2[i * CDIM], a2);
        }
        const float ck = coef[k];
        hcat1[(size_t)t * HC + k * CDIM + j] = (__bf16)(ck * silu_f(a1));
        hcat2[(size_t)t * HC + k * CDIM + j] = (__bf16)(ck * silu_f(a2));
    }

    // bias head: ybias[t, col] = c @ Wb + bb
    for (int q = 0; q < 4; ++q) {
        const int col = q * CDIM + j;
        float s = bb[col];
        for (int i = 0; i < CDIM; ++i) s = fmaf(cbuf[i], Wb[i * 1024 + col], s);
        ybias[(size_t)t * 1024 + col] = s;
    }
}

// ---------------------------------------------------------------------------
// Kernel 2: fused generate+apply for one Monarch layer.
// Grid (32 g-chunks, 32 token-tiles), 256 threads (8 wave32).
// Per WG: M[16 tok, 1024 cols] = hcat_tile[16,1024]bf16 @ Wbig[:, g*1024..][1024,1024]
//         (f32 weights converted to bf16 while staging to LDS), plus mixed bias,
//         then x1o[t,g,o] = sum_i x[t,g,i] * M[t, i*32+o].
// ---------------------------------------------------------------------------
#define A_STRIDE   1032                     // bf16 elems, bank-rotated (2064B = 516 dw)
#define A_OFF      0
#define A_BYTES    (16 * A_STRIDE * 2)      // 33024
#define BT_OFF     A_BYTES
#define BT_STRIDE  40                       // bf16 elems (80B = 20 dw rotation)
#define BT_BYTES   (1024 * BT_STRIDE * 2)   // 81920
#define M_OFF      BT_OFF                   // f32 epilogue buffer reuses B region
#define M_STRIDE   1028
#define BIAS_OFF   (BT_OFF + BT_BYTES)      // 114944
#define BIAS_BYTES (KEXP * 1024 * 4)        // 16384
#define X_OFF      (BIAS_OFF + BIAS_BYTES)  // 131328
#define X_BYTES    (16 * 32 * 4)            // 2048
#define CF_OFF     (X_OFF + X_BYTES)        // 133376
#define SMEM_TOTAL (CF_OFF + 16 * KEXP * 4) // 133632 bytes (< 160KB -> 2 WG / WGP)

__global__ __launch_bounds__(256) void hmx_gen_apply(
    const __bf16* __restrict__ hcat,   // [T, 1024] bf16
    const float*  __restrict__ Wbig,   // [1024, 32768] f32 (W{1,2}b flattened)
    const float*  __restrict__ bbig,   // [4, 32768] f32   (b{1,2}b)
    const float*  __restrict__ coeffs, // [T, 4]
    const float*  __restrict__ xin,    // [T, 1024] (layer1: x; layer2: transposed x1o)
    float*        __restrict__ xout,   // layer1: x2ws (transposed); layer2: final out
    const float*  __restrict__ ratio,  // [T]   (final stage only)
    const float*  __restrict__ ybias,  // [T,1024] (final stage only)
    int final_stage)
{
    extern __shared__ char smem[];
    __bf16* Asm = (__bf16*)(smem + A_OFF);
    __bf16* Bt  = (__bf16*)(smem + BT_OFF);
    float*  Mb  = (float*)(smem + M_OFF);
    float*  Bs  = (float*)(smem + BIAS_OFF);
    float*  Xt  = (float*)(smem + X_OFF);
    float*  Cf  = (float*)(smem + CF_OFF);

    const int g   = blockIdx.x;       // Monarch group / 1024-col chunk
    const int t0  = blockIdx.y * 16;  // token tile base
    const int tid = threadIdx.x;

    // ---- one-time staging: A tile (bf16), x tile, bias chunk, coeffs ----
    {
        const uint32_t* hs = (const uint32_t*)(hcat + (size_t)t0 * HC);
        for (int e = tid; e < 16 * 512; e += 256) {
            const int r = e >> 9, u = e & 511;
            ((uint32_t*)(Asm + r * A_STRIDE))[u] = hs[r * 512 + u];
        }
        for (int e = tid; e < 512; e += 256) {
            const int r = e >> 5, i = e & 31;
            Xt[e] = xin[(size_t)(t0 + r) * 1024 + g * 32 + i];
        }
        for (int e = tid; e < 1024; e += 256) {      // 4 rows x 256 float4
            const int k = e >> 8, c4 = e & 255;
            ((float4*)Bs)[e] = *((const float4*)(bbig + (size_t)k * DBIG + g * 1024) + c4);
        }
        if (tid < 16 * KEXP) Cf[tid] = coeffs[t0 * KEXP + tid];
    }

    const int wv   = tid >> 5;
    const int lane = tid & 31;
    const int nl   = lane & 15;
    const int half = lane >> 4;

    v8f acc[8];
    const v8f zero = {0.f, 0.f, 0.f, 0.f, 0.f, 0.f, 0.f, 0.f};
#pragma unroll
    for (int nt = 0; nt < 8; ++nt) acc[nt] = zero;

    const __bf16* Arow = Asm + nl * A_STRIDE;

    // ---- K loop: 1024 in steps of 32 (bf16 WMMA K) ----
    for (int kk = 0; kk < 32; ++kk) {
        const int k0 = kk * 32;
        __syncthreads();
        // stage B tile [32 k x 1024 n] f32 -> bf16, transposed into Bt[n][k];
        // branch-free speculative prefetch of the next k-tile (clamped in-bounds).
        const int kpf = (kk < 31 ? kk + 1 : 31) * 32;
        for (int e = tid; e < 8192; e += 256) {
            const int row = e >> 8, c4 = e & 255;
            const float4* src = (const float4*)(Wbig + (size_t)(k0 + row) * DBIG + g * 1024) + c4;
            const float4 w = *src;
            const int n = c4 * 4;
            Bt[(n + 0) * BT_STRIDE + row] = (__bf16)w.x;
            Bt[(n + 1) * BT_STRIDE + row] = (__bf16)w.y;
            Bt[(n + 2) * BT_STRIDE + row] = (__bf16)w.z;
            Bt[(n + 3) * BT_STRIDE + row] = (__bf16)w.w;
            __builtin_prefetch(
                (const void*)(Wbig + (size_t)(kpf + row) * DBIG + g * 1024 + c4 * 4), 0, 1);
        }
        __syncthreads();

        // A fragment (16x32 bf16, ISA layout: half selects K+8 / K+24 groups)
        const v8bf alo = *(const v8bf*)(Arow + k0 + 8 * half);
        const v8bf ahi = *(const v8bf*)(Arow + k0 + 16 + 8 * half);
        const v16bf afrag = __builtin_shufflevector(alo, ahi,
            0, 1, 2, 3, 4, 5, 6, 7, 8, 9, 10, 11, 12, 13, 14, 15);

        // Batch all 8 B-fragment loads first so the 16 ds_load_b128 are in
        // flight together; WMMAs then drain them with partial dscnt waits
        // instead of a full s_wait_dscnt 0 per tile.
        v16bf bfrag[8];
#pragma unroll
        for (int nt = 0; nt < 8; ++nt) {
            const int n = wv * 128 + nt * 16 + nl;
            const __bf16* bp = Bt + n * BT_STRIDE + 8 * half;
            const v8bf blo = *(const v8bf*)bp;
            const v8bf bhi = *(const v8bf*)(bp + 16);
            bfrag[nt] = __builtin_shufflevector(blo, bhi,
                0, 1, 2, 3, 4, 5, 6, 7, 8, 9, 10, 11, 12, 13, 14, 15);
        }
#pragma unroll
        for (int nt = 0; nt < 8; ++nt) {
            acc[nt] = __builtin_amdgcn_wmma_f32_16x16x32_bf16(
                false, afrag, false, bfrag[nt], (short)0, acc[nt], false, false);
        }
    }
    __syncthreads();   // all waves done with Bt before reusing region as Mb

    // ---- epilogue: add coefficient-mixed bias, spill M chunk to LDS ----
#pragma unroll
    for (int nt = 0; nt < 8; ++nt) {
        const int n = wv * 128 + nt * 16 + nl;
#pragma unroll
        for (int v = 0; v < 8; ++v) {
            const int m = v + 8 * half;    // token within tile (C/D f32 layout)
            float val = acc[nt][v];
#pragma unroll
            for (int k = 0; k < KEXP; ++k)
                val = fmaf(Cf[m * KEXP + k], Bs[k * 1024 + n], val);
            Mb[m * M_STRIDE + n] = val;
        }
    }
    __syncthreads();

    // ---- Monarch contraction: xo[t,g,o] = sum_i Xt[t,i] * M[t, i*32+o] ----
    for (int e = tid; e < 512; e += 256) {
        const int m = e >> 5, o = e & 31;
        const float* Mr = Mb + m * M_STRIDE;
        const float* xr = Xt + m * 32;
        float s = 0.f;
#pragma unroll 8
        for (int i = 0; i < 32; ++i) s = fmaf(xr[i], Mr[i * 32 + o], s);
        const int t = t0 + m;
        if (!final_stage) {
            xout[(size_t)t * 1024 + o * 32 + g] = s;   // store transposed for layer 2
        } else {
            xout[(size_t)t * 1024 + g * 32 + o] =
                fmaf(s, ratio[t], ybias[(size_t)t * 1024 + g * 32 + o]);
        }
    }
}

// ---------------------------------------------------------------------------
// Host launcher
// ---------------------------------------------------------------------------
extern "C" void kernel_launch(void* const* d_in, const int* in_sizes, int n_in,
                              void* d_out, int out_size, void* d_ws, size_t ws_size,
                              hipStream_t stream) {
    (void)in_sizes; (void)n_in; (void)out_size; (void)ws_size;

    const float* x   = (const float*)d_in[0];
    const float* Wc  = (const float*)d_in[1];
    const float* bc  = (const float*)d_in[2];
    const float* W1a = (const float*)d_in[3];
    const float* b1a = (const float*)d_in[4];
    const float* W1b = (const float*)d_in[5];
    const float* b1b = (const float*)d_in[6];
    const float* W2a = (const float*)d_in[7];
    const float* b2a = (const float*)d_in[8];
    const float* W2b = (const float*)d_in[9];
    const float* b2b = (const float*)d_in[10];
    const float* Wm1 = (const float*)d_in[11];
    const float* bm1 = (const float*)d_in[12];
    const float* Wm2 = (const float*)d_in[13];
    const float* bm2 = (const float*)d_in[14];
    const float* Wb  = (const float*)d_in[15];
    const float* bb  = (const float*)d_in[16];
    const float* Wr  = (const float*)d_in[17];
    const float* br  = (const float*)d_in[18];
    float* out = (float*)d_out;

    // workspace carve-up (~6 MB)
    char* ws = (char*)d_ws;
    __bf16* hcat1 = (__bf16*)(ws);                                   // 1 MB
    __bf16* hcat2 = (__bf16*)(ws + (1u << 20));                      // 1 MB
    float*  coeffs = (float*)(ws + (2u << 20));                      // 8 KB
    float*  ratio  = (float*)(ws + (2u << 20) + 8192);               // 2 KB
    float*  ybias  = (float*)(ws + (2u << 20) + 8192 + 2048);        // 2 MB
    float*  x2ws   = (float*)(ws + (2u << 20) + 8192 + 2048 +
                              (size_t)T_TOKENS * 1024 * 4);          // 2 MB

    hmx_prep<<<T_TOKENS, 256, 0, stream>>>(
        x, Wc, bc, W1a, b1a, W2a, b2a, Wm1, bm1, Wm2, bm2,
        Wb, bb, Wr, br, hcat1, hcat2, coeffs, ratio, ybias);

    dim3 grid(32, 32);  // (g-chunk, token-tile)
    hmx_gen_apply<<<grid, 256, SMEM_TOTAL, stream>>>(
        hcat1, W1b, b1b, coeffs, x, x2ws, nullptr, nullptr, 0);
    hmx_gen_apply<<<grid, 256, SMEM_TOTAL, stream>>>(
        hcat2, W2b, b2b, coeffs, x2ws, out, ratio, ybias, 1);
}
